// CRFClassifier_21698174779443
// MI455X (gfx1250) — matrix operand order repaired and checked by reading, not compiled, not run on hardware
//
#include <hip/hip_runtime.h>

typedef __attribute__((ext_vector_type(16))) _Float16 v16h;
typedef __attribute__((ext_vector_type(8)))  float    v8f;

#define LBL 128
#define NB  16              // batches per workgroup (WMMA N dimension)
#define NEG_INF_F (-10000.0f)
#define LOG2E_F 1.44269504088896340736f
#define LN2_F   0.69314718055994530942f

__global__ __launch_bounds__(256) void crf_forward_kernel(
    const float* __restrict__ xg,        // (B, T, L) fp32
    const float* __restrict__ trans,     // (L, L)    fp32
    const int*   __restrict__ lengths,   // (B,)      int32
    float*       __restrict__ out,       // (B,)      fp32
    int T)
{
  __shared__ float    fv[LBL][NB + 1];     // padded to dodge bank conflicts
  __shared__ _Float16 efT[NB][LBL + 4];    // transposed exp(fv - m), f16
  __shared__ float    xs[NB][LBL];         // x[:, t, :] staging
  __shared__ float    red[NB][16];
  __shared__ float    mrow[NB];
  __shared__ int      lens[NB];

  const int tid  = threadIdx.x;
  const int lane = tid & 31;
  const int wave = tid >> 5;               // 0..7: owns rows [16w, 16w+16)
  const int b0   = blockIdx.x * NB;

  const int ml = lane & 15;                // A: row-in-tile; B/C/D: batch column
  const int lh = lane >> 4;                // lane half

  // ---------------- init fv0, lengths ----------------
  {
    const int n   = tid & 15;
    const int seg = tid >> 4;              // 16 segments x 8 rows
    #pragma unroll
    for (int j = 0; j < 8; ++j) {
      int prev = seg * 8 + j;
      fv[prev][n] = (prev == 0) ? 0.0f : NEG_INF_F;   // START_IDX = 0
    }
    if (tid < NB) lens[tid] = lengths[b0 + tid];
  }

  // ---- A fragments: exp(trans) rows [16w,16w+16), f16, register-resident ----
  // ISA 16-bit A 16x32 layout: lane m=lane&15; VGPR v holds K =
  //   ((v<4) ? 2v : 8+2v) + 8*(lane>>4), per 32-wide K chunk.
  v16h aF[4];
  {
    const int row = wave * 16 + ml;
    #pragma unroll
    for (int kc = 0; kc < 4; ++kc) {
      #pragma unroll
      for (int v = 0; v < 8; ++v) {
        int k = kc * 32 + ((v < 4) ? 2 * v : 8 + 2 * v) + 8 * lh;
        aF[kc][2 * v]     = (_Float16)__builtin_amdgcn_exp2f(LOG2E_F * trans[row * LBL + k]);
        aF[kc][2 * v + 1] = (_Float16)__builtin_amdgcn_exp2f(LOG2E_F * trans[row * LBL + k + 1]);
      }
    }
  }
  __syncthreads();

  const int n16 = tid & 15;
  const int seg = tid >> 4;

  for (int t = 0; t < T; ++t) {
    // ---- stage x[:, t, :] into LDS (coalesced 16-lane groups) ----
    {
      const int n = tid >> 4;              // batch row 0..15
      const int l = tid & 15;
      const size_t base = ((size_t)(b0 + n) * (size_t)T + (size_t)t) * LBL;
      #pragma unroll
      for (int i = 0; i < 8; ++i)
        xs[n][l + 16 * i] = xg[base + l + 16 * i];
      if (t + 1 < T)
        __builtin_prefetch(&xg[base + LBL + l], 0, 0);  // global_prefetch_b8
    }

    // ---- per-batch max over prev: two-level reduction ----
    {
      float pm = -2.0e4f;
      #pragma unroll
      for (int j = 0; j < 8; ++j)
        pm = fmaxf(pm, fv[seg * 8 + j][n16]);
      red[n16][seg] = pm;
    }
    __syncthreads();
    if (tid < NB) {
      float mm = red[tid][0];
      #pragma unroll
      for (int j = 1; j < 16; ++j) mm = fmaxf(mm, red[tid][j]);
      mrow[tid] = mm;
    }
    __syncthreads();

    // ---- efT[n][prev] = (f16) 2^((fv - m) * log2e)  (arg <= 0, FTZ is fine) ----
    {
      const float mv = mrow[n16];
      #pragma unroll
      for (int j = 0; j < 8; ++j) {
        int prev = seg * 8 + j;
        efT[n16][prev] =
            (_Float16)__builtin_amdgcn_exp2f((fv[prev][n16] - mv) * LOG2E_F);
      }
    }
    __syncthreads();

    // ---- S[16x16] = exp(trans)_tile(16x128) x EF(128x16) via 4 WMMAs ----
    // B 32x16 f16 layout: lane n=lane&15; VGPR v holds K = ((v<4) ? 2v : 8+2v) + 8*(lane>>4).
    v8f c = {0.f, 0.f, 0.f, 0.f, 0.f, 0.f, 0.f, 0.f};
    #pragma unroll
    for (int kc = 0; kc < 4; ++kc) {
      v16h bF;
      #pragma unroll
      for (int v = 0; v < 8; ++v) {
        int k = kc * 32 + ((v < 4) ? 2 * v : 8 + 2 * v) + 8 * lh;
        bF[2 * v]     = efT[ml][k];
        bF[2 * v + 1] = efT[ml][k + 1];
      }
      c = __builtin_amdgcn_wmma_f32_16x16x32_f16(
              /*neg_a=*/false, aF[kc], /*neg_b=*/false, bF,
              /*c_mod=*/(short)0, c, /*reuse_a=*/false, /*reuse_b=*/false);
    }

    // ---- fv_new = (x_t + m) + ln2 * log2(S); mask-select (no divergence) ----
    // S in [~0.9, 141]: raw v_log_f32 is safe, no denorm rescue needed.
    // D layout: VGPR r -> M = r + 8*(lane>>4), N = lane&15.
    {
      const float mv     = mrow[ml];
      const bool  active = (t < lens[ml]);
      #pragma unroll
      for (int r = 0; r < 8; ++r) {
        int   nxt  = wave * 16 + r + 8 * lh;
        float fnew = fmaf(LN2_F, __builtin_amdgcn_logf(c[r]), xs[ml][nxt] + mv);
        float fold = fv[nxt][ml];
        fv[nxt][ml] = active ? fnew : fold;
      }
    }
    __syncthreads();
  }

  // ---------------- terminal: alpha = LSE_prev(fv + trans[STOP, prev]) ----
  if (tid < NB) {
    const float* tstop = trans + (LBL - 1) * LBL;   // STOP_IDX = L-1
    float mm = -3.0e38f;
    for (int prev = 0; prev < LBL; ++prev)
      mm = fmaxf(mm, fv[prev][tid] + tstop[prev]);
    float s = 0.0f;
    for (int prev = 0; prev < LBL; ++prev)
      s += __expf(fv[prev][tid] + tstop[prev] - mm);
    out[b0 + tid] = mm + __logf(s);
  }
}

extern "C" void kernel_launch(void* const* d_in, const int* in_sizes, int n_in,
                              void* d_out, int out_size, void* d_ws, size_t ws_size,
                              hipStream_t stream) {
  const float* x    = (const float*)d_in[0];   // (B, T, L)
  const float* tm   = (const float*)d_in[1];   // (L, L)
  const int*   len  = (const int*)d_in[2];     // (B,)
  float*       out  = (float*)d_out;           // (B,)

  const int B = in_sizes[2];
  const int T = in_sizes[0] / (B * LBL);

  dim3 grid(B / NB), block(256);
  crf_forward_kernel<<<grid, block, 0, stream>>>(x, tm, len, out, T);
}
